// Block_2302102471059
// MI455X (gfx1250) — compile-verified, matrix operand order checked by reading, not run on hardware
//
#include <hip/hip_runtime.h>
#include <hip/hip_bf16.h>
#include <stdint.h>

// ---------------------------------------------------------------------------
// Problem constants (match reference)
// ---------------------------------------------------------------------------
#define NN   65536
#define DD   512
#define GG   128
#define LL   512
#define E3D  1536   // 3*D

typedef __attribute__((ext_vector_type(16))) __bf16 v16bf;
typedef __attribute__((ext_vector_type(8)))  float  v8f;

union Frag { uint4 u[2]; v16bf v; };

__device__ __forceinline__ uint16_t f2bf(float f) {
    uint32_t u = __float_as_uint(f);
    uint32_t r = u + 0x7FFFu + ((u >> 16) & 1u);   // round-to-nearest-even
    return (uint16_t)(r >> 16);
}
__device__ __forceinline__ float bf2f(uint16_t h) {
    return __uint_as_float(((uint32_t)h) << 16);
}
__device__ __forceinline__ float sigmf(float x) {
    return 1.0f / (1.0f + __expf(-x));
}

// ---------------------------------------------------------------------------
// CDNA5 async global->LDS copy (ASYNCcnt-tracked) via inline asm (fallback
// path; verified to assemble in round 3).
// ---------------------------------------------------------------------------
#if defined(__gfx1250__)
#define ASYNC_LDS_OK 1
__device__ __forceinline__ void async_copy_b128(const void* g, void* l) {
    asm volatile("global_load_async_to_lds_b128 %0, %1, off"
                 :
                 : "v"((uint32_t)(uintptr_t)l), "v"((uint64_t)(uintptr_t)g)
                 : "memory");
}
#if __has_builtin(__builtin_amdgcn_s_wait_asynccnt)
#define WAIT_ASYNC(n) __builtin_amdgcn_s_wait_asynccnt(n)
#else
#define WAIT_ASYNC(n) asm volatile("s_wait_asynccnt %0" :: "n"(n) : "memory")
#endif
#else
#define ASYNC_LDS_OK 0
#endif

// ---------------------------------------------------------------------------
// CDNA5 Tensor Data Mover: one descriptor fetches a whole 2D bf16 tile
// (rows x cols, row stride = srcStride elements) into LDS, with hardware
// row padding (pad_interval=3 -> every 16 DWORDs, pad_amount=3 -> 4 DWORDs)
// reproducing our 40-half padded LDS rows. ISA ch.8 D# groups 0/1.
// ROCm 7.2 builtin = 5 args; amdgpu-toolchain (ships amd_gfx1250_TDM.h) = 6.
// ---------------------------------------------------------------------------
#if defined(__gfx1250__) && __has_builtin(__builtin_amdgcn_tensor_load_to_lds) && \
    __has_builtin(__builtin_amdgcn_s_wait_tensorcnt)
#define TDM_OK 1
#if __has_include(<hip/amd_detail/amd_gfx1250_TDM.h>)
#define TDM_SIX_ARGS 1
#else
#define TDM_SIX_ARGS 0
#endif
typedef __attribute__((ext_vector_type(4))) unsigned int v4u;
typedef __attribute__((ext_vector_type(8))) int v8i;
typedef __attribute__((ext_vector_type(4))) int v4i;

#define WAIT_TENSOR(n) __builtin_amdgcn_s_wait_tensorcnt((short)(n))

__device__ __forceinline__ void tdm_load_tile_pad(const uint16_t* g, uint32_t ldsAddr,
                                                  int rows, int cols, int srcStride) {
    uint64_t ga = (uint64_t)(uintptr_t)g;
    v4u g0;
    g0.x = 1u;                                   // count=1 (valid), user mode
    g0.y = ldsAddr;                              // lds_addr (bytes)
    g0.z = (uint32_t)ga;                         // global_addr[31:0]
    g0.w = (uint32_t)(ga >> 32) | (2u << 30);    // global_addr[56:32] | type=2
    v8i g1;
    g1[0] = (1 << 16)                            // data_size = 2 bytes
          | (1 << 20)                            // pad_enable
          | (3 << 22)                            // pad_interval: 16 DWORDs (one 64B row)
          | (3 << 25);                           // pad_amount: 4 DWORDs (16B)
    g1[1] = (srcStride & 0xFFFF) << 16;          // tensor_dim0[15:0]   (bits 48..63)
    g1[2] = ((srcStride >> 16) & 0xFFFF)         // tensor_dim0[31:16]
          | ((rows & 0xFFFF) << 16);             // tensor_dim1[15:0]
    g1[3] = ((rows >> 16) & 0xFFFF)              // tensor_dim1[31:16]
          | ((cols & 0xFFFF) << 16);             // tile_dim0
    g1[4] = rows & 0xFFFF;                       // tile_dim1 (tile_dim2 = 0)
    g1[5] = srcStride;                           // tensor_dim0_stride[31:0]
    g1[6] = 0;                                   // stride[47:32] / dim1_stride
    g1[7] = 0;
    v4i zz = {0, 0, 0, 0};
#if TDM_SIX_ARGS
    v8i z8 = {0, 0, 0, 0, 0, 0, 0, 0};
    __builtin_amdgcn_tensor_load_to_lds(g0, g1, zz, zz, z8, 0);
#else
    __builtin_amdgcn_tensor_load_to_lds(g0, g1, zz, zz, 0);
#endif
}
#else
#define TDM_OK 0
#endif

// ---------------------------------------------------------------------------
// Utility kernels
// ---------------------------------------------------------------------------
__global__ void zero_f32(float* __restrict__ p, size_t n) {
    size_t i = (size_t)blockIdx.x * blockDim.x + threadIdx.x;
    size_t stride = (size_t)gridDim.x * blockDim.x;
    for (; i < n; i += stride) p[i] = 0.0f;
}

__global__ void convert_bf16(const float* __restrict__ in, uint16_t* __restrict__ out, size_t n) {
    size_t i = (size_t)blockIdx.x * blockDim.x + threadIdx.x;
    size_t stride = (size_t)gridDim.x * blockDim.x;
    for (; i < n; i += stride) out[i] = f2bf(in[i]);
}

// Wt[n*Dm+k] = bf16(W[k*Dm+n])  (produce B^T layout for x @ W GEMMs)
__global__ void transpose_to_bf16(const float* __restrict__ W, uint16_t* __restrict__ Wt, int Dm) {
    int k = blockIdx.x * 16 + threadIdx.x;
    int n = blockIdx.y * 16 + threadIdx.y;
    if (k < Dm && n < Dm) Wt[(size_t)n * Dm + k] = f2bf(W[(size_t)k * Dm + n]);
}

// Edge mean-aggregation: agg[dst] += feats[src], deg[dst] += 1
__global__ __launch_bounds__(128)
void edge_agg(const float* __restrict__ feats, const int* __restrict__ esrc,
              const int* __restrict__ edst, float* __restrict__ agg,
              float* __restrict__ deg, int E) {
    for (int e = blockIdx.x; e < E; e += gridDim.x) {
        int s = esrc[e], d = edst[e];
        int e2 = e + gridDim.x;
        if (e2 < E) {
            int s2 = esrc[e2];
            __builtin_prefetch(feats + (size_t)s2 * DD + threadIdx.x * 4, 0, 0);
        }
        float4 v = ((const float4*)(feats + (size_t)s * DD))[threadIdx.x];
        float* ap = agg + (size_t)d * DD + threadIdx.x * 4;
        atomicAdd(ap + 0, v.x);
        atomicAdd(ap + 1, v.y);
        atomicAdd(ap + 2, v.z);
        atomicAdd(ap + 3, v.w);
        if (threadIdx.x == 0) atomicAdd(deg + d, 1.0f);
    }
}

// x_bf16 = bf16(agg / max(deg,1))
__global__ void finalize_agg(const float* __restrict__ agg, const float* __restrict__ deg,
                             uint16_t* __restrict__ x, size_t total) {
    size_t i = (size_t)blockIdx.x * blockDim.x + threadIdx.x;
    size_t stride = (size_t)gridDim.x * blockDim.x;
    for (; i < total; i += stride) {
        size_t n = i >> 9;               // /D
        float s = 1.0f / fmaxf(deg[n], 1.0f);
        x[i] = f2bf(agg[i] * s);
    }
}

// seqs[(l*G+g)] = med[seq_ids[g*L+l]]   (bf16 rows, 1024B each)
__global__ __launch_bounds__(128)
void gather_rows(const uint16_t* __restrict__ med, const int* __restrict__ seq_ids,
                 uint16_t* __restrict__ seqs) {
    int r = blockIdx.x;                  // r = l*G + g
    int l = r >> 7, g = r & (GG - 1);
    int node = seq_ids[g * LL + l];
    const uint2* src = (const uint2*)(med + (size_t)node * DD);
    uint2* dst = (uint2*)(seqs + (size_t)r * DD);
    dst[threadIdx.x] = src[threadIdx.x]; // 128 thr * 8B = 1024B
}

// ---------------------------------------------------------------------------
// Generic bf16 WMMA GEMM:  C[M,Nout] = A[M,K] * Bt[Nout,K]^T + bias, opt relu
// 128x128 tile per block, 256 threads (8 waves), TK=32, double-buffered LDS.
// Tile fill: TDM (2 descriptors/tile, wave 0, TENSORcnt) when available;
// else per-lane GLOBAL_LOAD_ASYNC_TO_LDS_B128 (ASYNCcnt); else sync copy.
// LDS rows padded 32->40 halves (80B = 20-bank stride, conflict-free frags).
// ---------------------------------------------------------------------------
#define TM  128
#define TN  128
#define TK  32
#define TKP 40   // padded LDS row stride (halves)

__global__ __launch_bounds__(256)
void gemm_bf16(const uint16_t* __restrict__ A, const uint16_t* __restrict__ Bt,
               const float* __restrict__ bias, void* __restrict__ Cout,
               int M, int Nout, int K, int relu, int outBf16) {
    __shared__ __align__(16) uint16_t As[2][TM][TKP];
    __shared__ __align__(16) uint16_t Bs[2][TN][TKP];

    const int tid  = threadIdx.x;
    const int lane = tid & 31;
    const int wave = tid >> 5;
    const int wm   = wave >> 2;     // 0..1  -> 64 rows each
    const int wn   = wave & 3;      // 0..3  -> 32 cols each
    const int m0   = blockIdx.y * TM;
    const int n0   = blockIdx.x * TN;

    const int lr = tid >> 1;
    const int lc = (tid & 1) * 16;

    v8f acc[4][2];
    #pragma unroll
    for (int i = 0; i < 4; i++)
        #pragma unroll
        for (int j = 0; j < 2; j++)
            #pragma unroll
            for (int e = 0; e < 8; e++) acc[i][j][e] = 0.0f;

#if TDM_OK
    auto issue_tile = [&](int kt, int buf) {
        if (wave == 0) {
            const uint16_t* ag = A  + (size_t)m0 * K + (size_t)kt * TK;
            const uint16_t* bg = Bt + (size_t)n0 * K + (size_t)kt * TK;
            tdm_load_tile_pad(ag, (uint32_t)(uintptr_t)&As[buf][0][0], TM, TK, K);
            tdm_load_tile_pad(bg, (uint32_t)(uintptr_t)&Bs[buf][0][0], TN, TK, K);
        }
    };
#elif ASYNC_LDS_OK
    auto issue_tile = [&](int kt, int buf) {
        const size_t ka = (size_t)kt * TK;
        const uint16_t* ag = A  + (size_t)(m0 + lr) * K + ka + lc;
        const uint16_t* bg = Bt + (size_t)(n0 + lr) * K + ka + lc;
        async_copy_b128(ag,     &As[buf][lr][lc]);
        async_copy_b128(ag + 8, &As[buf][lr][lc + 8]);
        async_copy_b128(bg,     &Bs[buf][lr][lc]);
        async_copy_b128(bg + 8, &Bs[buf][lr][lc + 8]);
    };
#else
    auto issue_tile = [&](int kt, int buf) {
        const size_t ka = (size_t)kt * TK;
        const uint16_t* ag = A  + (size_t)(m0 + lr) * K + ka + lc;
        const uint16_t* bg = Bt + (size_t)(n0 + lr) * K + ka + lc;
        uint4 a0 = *(const uint4*)(ag);
        uint4 a1 = *(const uint4*)(ag + 8);
        uint4 b0 = *(const uint4*)(bg);
        uint4 b1 = *(const uint4*)(bg + 8);
        *(uint4*)&As[buf][lr][lc]     = a0;
        *(uint4*)&As[buf][lr][lc + 8] = a1;
        *(uint4*)&Bs[buf][lr][lc]     = b0;
        *(uint4*)&Bs[buf][lr][lc + 8] = b1;
    };
#endif

    auto compute = [&](int buf) {
        const int half = lane >> 4;
        const int mr   = lane & 15;
        Frag aF[4], bF[2];
        #pragma unroll
        for (int ms = 0; ms < 4; ms++) {
            const uint16_t* p = &As[buf][wm * 64 + ms * 16 + mr][8 * half];
            aF[ms].u[0] = *(const uint4*)(p);
            aF[ms].u[1] = *(const uint4*)(p + 16);
        }
        #pragma unroll
        for (int ns = 0; ns < 2; ns++) {
            const uint16_t* p = &Bs[buf][wn * 32 + ns * 16 + mr][8 * half];
            bF[ns].u[0] = *(const uint4*)(p);
            bF[ns].u[1] = *(const uint4*)(p + 16);
        }
        #pragma unroll
        for (int ms = 0; ms < 4; ms++)
            #pragma unroll
            for (int ns = 0; ns < 2; ns++)
                acc[ms][ns] = __builtin_amdgcn_wmma_f32_16x16x32_bf16(
                    false, aF[ms].v, false, bF[ns].v, (short)0, acc[ms][ns], false, false);
    };

    const int KT = K / TK;
    int buf = 0;
#if TDM_OK
    issue_tile(0, 0);                      // 2 TDM ops outstanding (wave 0)
    for (int kt = 0; kt < KT; kt++) {
        if (kt + 1 < KT) {
            issue_tile(kt + 1, buf ^ 1);   // <= 4 outstanding on wave 0
            if (wave == 0) WAIT_TENSOR(2); // tile kt landed (in-order per wave)
        } else {
            if (wave == 0) WAIT_TENSOR(0);
        }
        __syncthreads();                   // tile kt visible to all waves
        compute(buf);
        __syncthreads();                   // all reads of buf done before reuse
        buf ^= 1;
    }
#elif ASYNC_LDS_OK
    issue_tile(0, 0);
    for (int kt = 0; kt < KT; kt++) {
        if (kt + 1 < KT) {
            issue_tile(kt + 1, buf ^ 1);
            WAIT_ASYNC(4);
        } else {
            WAIT_ASYNC(0);
        }
        __syncthreads();
        compute(buf);
        __syncthreads();
        buf ^= 1;
    }
#else
    issue_tile(0, 0);
    __syncthreads();
    for (int kt = 0; kt < KT; kt++) {
        if (kt + 1 < KT) issue_tile(kt + 1, buf ^ 1);
        compute(buf);
        __syncthreads();
        buf ^= 1;
    }
#endif

    // store
    const int half = lane >> 4;
    const int nn   = lane & 15;
    #pragma unroll
    for (int ms = 0; ms < 4; ms++) {
        #pragma unroll
        for (int ns = 0; ns < 2; ns++) {
            int col = n0 + wn * 32 + ns * 16 + nn;
            float bv = bias ? bias[col] : 0.0f;
            #pragma unroll
            for (int j = 0; j < 8; j++) {
                int row = m0 + wm * 64 + ms * 16 + j + 8 * half;
                float v = acc[ms][ns][j] + bv;
                if (relu) v = fmaxf(v, 0.0f);
                if (outBf16)
                    ((uint16_t*)Cout)[(size_t)row * Nout + col] = f2bf(v);
                else
                    ((float*)Cout)[(size_t)row * Nout + col] = v;
            }
        }
    }
}

// ---------------------------------------------------------------------------
// Fused GRU recurrence over all 512 steps.
// grid = 8 blocks; block b owns sequences g = b*16 .. b*16+15 (no cross-block deps).
// 512 threads = 16 waves; wave w owns gate columns [32w,32w+32) for r/z/n,
// so gate math uses only its own WMMA accumulators. h kept in LDS (f32 + bf16).
// hB rows padded 512->520 halves (260-dword stride => conflict-free frags).
// ---------------------------------------------------------------------------
#define HBP 520

__global__ __launch_bounds__(512)
void gru_seq(const uint16_t* __restrict__ gi,    // [L][G][3D] bf16 (b_ih included)
             const uint16_t* __restrict__ Whh,   // [3D][D] bf16 (B^T layout)
             const float* __restrict__ bhh,      // [3D]
             const int* __restrict__ seq_ids,    // [G][L]
             uint16_t* __restrict__ ysn) {       // [N][D] bf16, scattered
    __shared__ __align__(16) float    hF[16][DD];   // 32KB
    __shared__ __align__(16) uint16_t hB[16][HBP];  // 16.25KB

    const int tid = threadIdx.x;
    for (int i = tid; i < 16 * HBP; i += 512) (&hB[0][0])[i] = 0;
    for (int i = tid; i < 16 * DD;  i += 512) (&hF[0][0])[i] = 0.0f;
    __syncthreads();

    const int lane = tid & 31;
    const int w    = tid >> 5;      // wave 0..15
    const int half = lane >> 4;
    const int nn   = lane & 15;
    const int g0   = blockIdx.x * 16;

    float bias[3][2];
    #pragma unroll
    for (int gate = 0; gate < 3; gate++)
        #pragma unroll
        for (int sub = 0; sub < 2; sub++)
            bias[gate][sub] = bhh[gate * DD + 32 * w + sub * 16 + nn];

    for (int l = 0; l < LL; l++) {
        if (l + 1 < LL) {
            const uint16_t* nxt = gi + ((size_t)(l + 1) * GG + g0) * E3D;
            __builtin_prefetch(nxt + (size_t)tid * 48, 0, 0);   // 512 thr * 96B = 48KB
        }

        v8f acc[3][2];
        #pragma unroll
        for (int gate = 0; gate < 3; gate++)
            #pragma unroll
            for (int sub = 0; sub < 2; sub++)
                #pragma unroll
                for (int e = 0; e < 8; e++) acc[gate][sub][e] = bias[gate][sub];

        const uint16_t* giBase = gi + ((size_t)l * GG + g0) * E3D;

        // gh = h @ Whh^T  (K = 512, 16 k-steps of 32)
        for (int kt = 0; kt < 16; kt++) {
            Frag aF;
            const uint16_t* ap = &hB[nn][kt * 32 + 8 * half];
            aF.u[0] = *(const uint4*)(ap);
            aF.u[1] = *(const uint4*)(ap + 16);
            #pragma unroll
            for (int gate = 0; gate < 3; gate++) {
                #pragma unroll
                for (int sub = 0; sub < 2; sub++) {
                    int e = gate * DD + 32 * w + sub * 16 + nn;
                    const uint16_t* bp = Whh + (size_t)e * DD + kt * 32 + 8 * half;
                    Frag bF;
                    bF.u[0] = *(const uint4*)(bp);
                    bF.u[1] = *(const uint4*)(bp + 16);
                    acc[gate][sub] = __builtin_amdgcn_wmma_f32_16x16x32_bf16(
                        false, aF.v, false, bF.v, (short)0, acc[gate][sub], false, false);
                }
            }
        }
        __syncthreads();   // all waves done reading hB before it is rewritten

        int node[8];
        #pragma unroll
        for (int j = 0; j < 8; j++)
            node[j] = seq_ids[(g0 + j + 8 * half) * LL + l];

        #pragma unroll
        for (int sub = 0; sub < 2; sub++) {
            int c = 32 * w + sub * 16 + nn;
            #pragma unroll
            for (int j = 0; j < 8; j++) {
                int m = j + 8 * half;     // local sequence index 0..15
                const uint16_t* gr = giBase + (size_t)m * E3D;
                float ir  = bf2f(gr[c]);
                float iz  = bf2f(gr[DD + c]);
                float inn = bf2f(gr[2 * DD + c]);
                float r  = sigmf(ir + acc[0][sub][j]);
                float z  = sigmf(iz + acc[1][sub][j]);
                float nv = tanhf(inn + r * acc[2][sub][j]);
                float hnew = (1.0f - z) * nv + z * hF[m][c];
                hF[m][c] = hnew;
                hB[m][c] = f2bf(hnew);
                ysn[(size_t)node[j] * DD + c] = f2bf(hnew);
            }
        }
        __syncthreads();
    }
}

// ---------------------------------------------------------------------------
// Workspace layout (bytes) — aggressively reused, total ~390 MB
// ---------------------------------------------------------------------------
static constexpr size_t OFF_AGG  = 0;                    // f32  N*D   (134217728)
static constexpr size_t OFF_X    = 134217728;            // bf16 N*D   (67108864)
static constexpr size_t OFF_GI   = 0;                    // bf16 N*3D  (201326592) overlays AGG+X
static constexpr size_t OFF_DEG  = 201326592;            // f32  N     (262144)
static constexpr size_t OFF_C1   = 201588736;            // bf16 N*D   (conv out; later ysn)
static constexpr size_t OFF_YSN  = OFF_C1;
static constexpr size_t OFF_MED  = 268697600;            // bf16 N*D
static constexpr size_t OFF_SEQS = 335806464;            // bf16 N*D
static constexpr size_t OFF_WCT  = 402915328;            // bf16 D*D
static constexpr size_t OFF_WF1T = 403439616;            // bf16 D*D
static constexpr size_t OFF_WF2T = 403963904;            // bf16 D*D
static constexpr size_t OFF_WIH  = 404488192;            // bf16 3D*D
static constexpr size_t OFF_WHH  = 406061056;            // bf16 3D*D  (end 407633920)

extern "C" void kernel_launch(void* const* d_in, const int* in_sizes, int n_in,
                              void* d_out, int out_size, void* d_ws, size_t ws_size,
                              hipStream_t stream) {
    const float* in_feats = (const float*)d_in[0];
    const int*   esrc     = (const int*)d_in[1];
    const int*   edst     = (const int*)d_in[2];
    const int*   seq_ids  = (const int*)d_in[3];
    const float* W_conv   = (const float*)d_in[4];
    const float* b_conv   = (const float*)d_in[5];
    const float* W_ff1    = (const float*)d_in[6];
    const float* b_ff1    = (const float*)d_in[7];
    const float* W_ih     = (const float*)d_in[8];
    const float* W_hh     = (const float*)d_in[9];
    const float* b_ih     = (const float*)d_in[10];
    const float* b_hh     = (const float*)d_in[11];
    const float* W_ff2    = (const float*)d_in[12];
    const float* b_ff2    = (const float*)d_in[13];
    const int E = in_sizes[1];

    char* ws = (char*)d_ws;
    float*    agg  = (float*)(ws + OFF_AGG);
    float*    deg  = (float*)(ws + OFF_DEG);
    uint16_t* xbf  = (uint16_t*)(ws + OFF_X);
    uint16_t* gibf = (uint16_t*)(ws + OFF_GI);
    uint16_t* c1   = (uint16_t*)(ws + OFF_C1);
    uint16_t* ysn  = (uint16_t*)(ws + OFF_YSN);
    uint16_t* med  = (uint16_t*)(ws + OFF_MED);
    uint16_t* seqs = (uint16_t*)(ws + OFF_SEQS);
    uint16_t* Wct  = (uint16_t*)(ws + OFF_WCT);
    uint16_t* Wf1t = (uint16_t*)(ws + OFF_WF1T);
    uint16_t* Wf2t = (uint16_t*)(ws + OFF_WF2T);
    uint16_t* Wih  = (uint16_t*)(ws + OFF_WIH);
    uint16_t* Whh  = (uint16_t*)(ws + OFF_WHH);

    const size_t ND = (size_t)NN * DD;

    // 1) mean aggregation over incoming edges
    zero_f32<<<65536, 256, 0, stream>>>(agg, ND);
    zero_f32<<<256, 256, 0, stream>>>(deg, (size_t)NN);
    edge_agg<<<131072, 128, 0, stream>>>(in_feats, esrc, edst, agg, deg, E);
    finalize_agg<<<65536, 256, 0, stream>>>(agg, deg, xbf, ND);

    // 2) weight prep (bf16, B^T layout where needed)
    dim3 tb(16, 16), tg(DD / 16, DD / 16);
    transpose_to_bf16<<<tg, tb, 0, stream>>>(W_conv, Wct, DD);
    transpose_to_bf16<<<tg, tb, 0, stream>>>(W_ff1, Wf1t, DD);
    transpose_to_bf16<<<tg, tb, 0, stream>>>(W_ff2, Wf2t, DD);
    convert_bf16<<<1536, 256, 0, stream>>>(W_ih, Wih, (size_t)E3D * DD);
    convert_bf16<<<1536, 256, 0, stream>>>(W_hh, Whh, (size_t)E3D * DD);

    // 3) conv linear + ff1(relu)
    gemm_bf16<<<dim3(DD / TN, NN / TM), 256, 0, stream>>>(xbf, Wct, b_conv, c1,
                                                          NN, DD, DD, 0, 1);
    gemm_bf16<<<dim3(DD / TN, NN / TM), 256, 0, stream>>>(c1, Wf1t, b_ff1, med,
                                                          NN, DD, DD, 1, 1);

    // 4) gather into [L,G,D], input projection gi = seqs @ W_ih^T + b_ih
    gather_rows<<<NN, 128, 0, stream>>>(med, seq_ids, seqs);
    gemm_bf16<<<dim3(E3D / TN, NN / TM), 256, 0, stream>>>(seqs, Wih, b_ih, gibf,
                                                           NN, E3D, DD, 0, 1);

    // 5) fused 512-step GRU recurrence, scattering h_t to node order
    gru_seq<<<GG / 16, 512, 0, stream>>>(gibf, Whh, b_hh, seq_ids, ysn);

    // 6) ff2 -> f32 output
    gemm_bf16<<<dim3(DD / TN, NN / TM), 256, 0, stream>>>(ysn, Wf2t, b_ff2, d_out,
                                                          NN, DD, DD, 0, 0);
}
